// SpikeQwen3Attention_10058813407415
// MI455X (gfx1250) — compile-verified, matrix operand order checked by reading, not run on hardware
//
#include <hip/hip_runtime.h>
#include <hip/hip_bf16.h>
#include <stdint.h>

// ---------------- problem constants ----------------
#define B_   2
#define S_   2048
#define HID_ 2048
#define H_   16
#define KV_  8
#define D_   128
#define MROWS (B_ * S_)                 // 4096
#define NQKV  (H_ * D_ + 2 * KV_ * D_) // 4096

typedef _Float16 half16 __attribute__((ext_vector_type(16)));
typedef _Float16 half4  __attribute__((ext_vector_type(4)));
typedef float    float8 __attribute__((ext_vector_type(8)));
typedef uint32_t u32x4  __attribute__((ext_vector_type(4)));
typedef int      i32x4  __attribute__((ext_vector_type(4)));

// ---------------- CDNA5 async global->LDS copy (ASYNCcnt path) ----------------
#if defined(__AMDGCN__) && __has_builtin(__builtin_amdgcn_global_load_async_to_lds_b128)
#define HAVE_ASYNC_LDS 1
#else
#define HAVE_ASYNC_LDS 0
#endif

#if HAVE_ASYNC_LDS
typedef __attribute__((address_space(1))) i32x4* gv4p;  // global int4*
typedef __attribute__((address_space(3))) i32x4* lv4p;  // LDS int4*
#endif

__device__ __forceinline__ void cp16_g2lds(const _Float16* gsrc, _Float16* ldst) {
#if HAVE_ASYNC_LDS
  // Build AS-qualified pointers from integers: flat LDS addresses carry the
  // LDS offset in addr[31:0] (aperture mapping), and global flat == AS1.
  gv4p g = (gv4p)(size_t)(const void*)gsrc;
  lv4p l = (lv4p)(size_t)(uint32_t)(size_t)(void*)ldst;
  __builtin_amdgcn_global_load_async_to_lds_b128(g, l, 0, 0);
#else
  *(u32x4*)ldst = *(const u32x4*)gsrc;
#endif
}

__device__ __forceinline__ void wait_async_lds() {
#if HAVE_ASYNC_LDS
#if __has_builtin(__builtin_amdgcn_s_wait_asynccnt)
  __builtin_amdgcn_s_wait_asynccnt(0);
#else
  asm volatile("s_wait_asynccnt 0x0" ::: "memory");
#endif
#endif
}

// ---------------- WMMA helpers ----------------
__device__ __forceinline__ float8 wmma_f16(half16 a, half16 b, float8 c) {
  // D = A(16x32 f16) * B(32x16 f16) + C(16x16 f32)
  return __builtin_amdgcn_wmma_f32_16x16x32_f16(false, a, false, b, (short)0, c,
                                                false, false);
}

// Load one 16x32 f16 A/B fragment from row-major storage with given stride.
// Per CDNA5 ISA (CDNA5 ISA): lane<16 -> row=lane, K {0..7, 16..23};
// lane>=16 -> row=lane-16, K {8..15, 24..31}. Two 16B loads per lane.
__device__ __forceinline__ half16 frag16(const _Float16* p, int stride, int lane) {
  const _Float16* q = p + (lane & 15) * stride + ((lane & 16) ? 8 : 0);
  half16 f;
  ((u32x4*)&f)[0] = *(const u32x4*)(q);
  ((u32x4*)&f)[1] = *(const u32x4*)(q + 16);
  return f;
}

__device__ __forceinline__ float8 zero8() {
  float8 z = {0.f, 0.f, 0.f, 0.f, 0.f, 0.f, 0.f, 0.f};
  return z;
}

// ---------------- kernel 1a: f32 -> f16 convert ----------------
__global__ __launch_bounds__(256) void cvt_f32_f16(const float* __restrict__ in,
                                                   _Float16* __restrict__ out,
                                                   size_t n) {
  size_t i = ((size_t)blockIdx.x * blockDim.x + threadIdx.x) * 4;
  if (i + 3 < n) {
    float4 v = *(const float4*)(in + i);
    half4 h = {(_Float16)v.x, (_Float16)v.y, (_Float16)v.z, (_Float16)v.w};
    *(half4*)(out + i) = h;
  }
}

// ---------------- kernel 1b: transpose (KxN) f32 -> (NxK) f16 ----------------
__global__ __launch_bounds__(256) void transpose_cvt(const float* __restrict__ in,
                                                     _Float16* __restrict__ out,
                                                     int K, int N) {
  size_t idx = (size_t)blockIdx.x * blockDim.x + threadIdx.x;
  if (idx < (size_t)K * N) {
    int k = (int)(idx / N);
    int n = (int)(idx % N);
    out[(size_t)n * K + k] = (_Float16)in[idx];
  }
}

// ---------------- kernel 2/5: tiled f16 WMMA GEMM ----------------
// C[M x N] (f32) = A[M x K] (f16, row-major) * B (f16, stored N-major: Bt[N x K])
// Block: 256 threads = 8 waves; block tile 128x128; wave tile 32x64 (2x4 WMMA);
// K-step 64 => 16 WMMAs per wave per barrier interval, 12 fragment loads.
__global__ __launch_bounds__(256) void gemm_f16_wmma(
    const _Float16* __restrict__ A, const _Float16* __restrict__ Bt,
    float* __restrict__ C, int M, int N, int K) {
  __shared__ _Float16 As[128 * 64];   // 16 KB
  __shared__ _Float16 Bs[128 * 64];   // 16 KB

  const int m0 = blockIdx.y * 128;
  const int n0 = blockIdx.x * 128;
  const int tid = threadIdx.x;
  const int wave = tid >> 5, lane = tid & 31;
  const int wm = (wave >> 1) * 32;    // 0/32/64/96
  const int wn = (wave & 1) * 64;     // 0/64

  float8 acc[2][4];
#pragma unroll
  for (int i = 0; i < 2; i++)
#pragma unroll
    for (int j = 0; j < 4; j++) acc[i][j] = zero8();

  const int r  = tid >> 1;            // 0..127: tile row this thread stages
  const int hh = (tid & 1) * 32;      // element offset within the 64-wide K slice

  for (int k0 = 0; k0 < K; k0 += 64) {
    __syncthreads();
    {
      const _Float16* sa = A  + (size_t)(m0 + r) * K + k0 + hh;
      const _Float16* sb = Bt + (size_t)(n0 + r) * K + k0 + hh;
      _Float16* da = As + r * 64 + hh;
      _Float16* db = Bs + r * 64 + hh;
#pragma unroll
      for (int i = 0; i < 4; i++) {   // 4 x 16B = 64B per matrix per thread
        cp16_g2lds(sa + i * 8, da + i * 8);
        cp16_g2lds(sb + i * 8, db + i * 8);
      }
      if (k0 + 64 < K) {              // gfx1250 global_prefetch_b8 for next slice
        __builtin_prefetch(sa + 64, 0, 1);
        __builtin_prefetch(sb + 64, 0, 1);
      }
    }
    wait_async_lds();
    __syncthreads();

#pragma unroll
    for (int kk = 0; kk < 2; kk++) {  // two 32-deep WMMA K-steps
      half16 a0 = frag16(As + (size_t)(wm)      * 64 + kk * 32, 64, lane);
      half16 a1 = frag16(As + (size_t)(wm + 16) * 64 + kk * 32, 64, lane);
#pragma unroll
      for (int j = 0; j < 4; j++) {
        half16 bj = frag16(Bs + (size_t)(wn + j * 16) * 64 + kk * 32, 64, lane);
        acc[0][j] = wmma_f16(a0, bj, acc[0][j]);
        acc[1][j] = wmma_f16(a1, bj, acc[1][j]);
      }
    }
  }

  // C/D layout: VGPR e holds row (e | (lane>=16 ? 8:0)), col = lane&15.
  const int lr = (lane & 16) ? 8 : 0;
  const int cb = lane & 15;
#pragma unroll
  for (int i = 0; i < 2; i++)
#pragma unroll
    for (int j = 0; j < 4; j++)
#pragma unroll
      for (int e = 0; e < 8; e++)
        C[(size_t)(m0 + wm + i * 16 + lr + e) * N + (n0 + wn + j * 16 + cb)] =
            acc[i][j][e];
}

// ---------------- kernel 3: RMSNorm + RoPE + GQA repack ----------------
// QKV f32 row layout: [q(2048) | k(1024) | v(1024)] per (b,s).
// Writes f16 Q[b][h][s][d] (scaled by 1/sqrt(D)), K[b][kv][s][d], V[b][kv][s][d].
__global__ __launch_bounds__(128) void qkv_post(
    const float* __restrict__ QKV, const float* __restrict__ qnw,
    const float* __restrict__ knw, const int* __restrict__ positions,
    _Float16* __restrict__ Qh, _Float16* __restrict__ Kh,
    _Float16* __restrict__ Vh) {
  const int s = blockIdx.x, b = blockIdx.y;
  const int tid = threadIdx.x, wave = tid >> 5, lane = tid & 31;
  const float* row = QKV + ((size_t)b * S_ + s) * NQKV;
  const float pos = (float)positions[s];

  // inv_freq[j] = THETA^(-j/64); lanes hold d = {lane, lane+32, lane+64, lane+96}.
  const float kfac = 0.21586735593695742f;  // ln(1e6)/64
  const float a0 = pos * __expf(-(float)lane * kfac);
  const float a1 = pos * __expf(-(float)(lane + 32) * kfac);
  const float cs0 = cosf(a0), sn0 = sinf(a0);
  const float cs1 = cosf(a1), sn1 = sinf(a1);

  // -------- Q heads (4 per wave) --------
  const float wq0 = qnw[lane], wq1 = qnw[lane + 32];
  const float wq2 = qnw[lane + 64], wq3 = qnw[lane + 96];
  for (int h = wave; h < H_; h += 4) {
    const float* px = row + h * D_;
    float x0 = px[lane], x1 = px[lane + 32], x2 = px[lane + 64], x3 = px[lane + 96];
    float ss = x0 * x0 + x1 * x1 + x2 * x2 + x3 * x3;
#pragma unroll
    for (int off = 1; off < 32; off <<= 1) ss += __shfl_xor(ss, off);
    float rn = rsqrtf(ss * (1.0f / 128.0f) + 1e-6f);
    x0 *= rn * wq0; x1 *= rn * wq1; x2 *= rn * wq2; x3 *= rn * wq3;
    const float qs = 0.08838834764831845f;  // 1/sqrt(128) folded into Q
    float y0 = (x0 * cs0 - x2 * sn0) * qs;
    float y1 = (x1 * cs1 - x3 * sn1) * qs;
    float y2 = (x2 * cs0 + x0 * sn0) * qs;
    float y3 = (x3 * cs1 + x1 * sn1) * qs;
    _Float16* dst = Qh + (((size_t)b * H_ + h) * S_ + s) * D_;
    dst[lane] = (_Float16)y0; dst[lane + 32] = (_Float16)y1;
    dst[lane + 64] = (_Float16)y2; dst[lane + 96] = (_Float16)y3;
  }
  // -------- K heads (2 per wave) --------
  const float wk0 = knw[lane], wk1 = knw[lane + 32];
  const float wk2 = knw[lane + 64], wk3 = knw[lane + 96];
  for (int kv = wave; kv < KV_; kv += 4) {
    const float* px = row + H_ * D_ + kv * D_;
    float x0 = px[lane], x1 = px[lane + 32], x2 = px[lane + 64], x3 = px[lane + 96];
    float ss = x0 * x0 + x1 * x1 + x2 * x2 + x3 * x3;
#pragma unroll
    for (int off = 1; off < 32; off <<= 1) ss += __shfl_xor(ss, off);
    float rn = rsqrtf(ss * (1.0f / 128.0f) + 1e-6f);
    x0 *= rn * wk0; x1 *= rn * wk1; x2 *= rn * wk2; x3 *= rn * wk3;
    float y0 = x0 * cs0 - x2 * sn0;
    float y1 = x1 * cs1 - x3 * sn1;
    float y2 = x2 * cs0 + x0 * sn0;
    float y3 = x3 * cs1 + x1 * sn1;
    _Float16* dst = Kh + (((size_t)b * KV_ + kv) * S_ + s) * D_;
    dst[lane] = (_Float16)y0; dst[lane + 32] = (_Float16)y1;
    dst[lane + 64] = (_Float16)y2; dst[lane + 96] = (_Float16)y3;
  }
  // -------- V heads (copy/convert) --------
  for (int kv = wave; kv < KV_; kv += 4) {
    const float* px = row + H_ * D_ + KV_ * D_ + kv * D_;
    _Float16* dst = Vh + (((size_t)b * KV_ + kv) * S_ + s) * D_;
    dst[lane] = (_Float16)px[lane];
    dst[lane + 32] = (_Float16)px[lane + 32];
    dst[lane + 64] = (_Float16)px[lane + 64];
    dst[lane + 96] = (_Float16)px[lane + 96];
  }
}

// ---------------- kernel 4: flash attention (causal, GQA) ----------------
// Grid: (S/64, H, B). Block 128 = 4 waves; each wave owns 16 q rows, so
// online-softmax state (m,l) and the O accumulators never cross waves.
__global__ __launch_bounds__(128) void flash_attn(
    const _Float16* __restrict__ Qh, const _Float16* __restrict__ Kh,
    const _Float16* __restrict__ Vh, _Float16* __restrict__ Oout) {
  const int qt = blockIdx.x, h = blockIdx.y, b = blockIdx.z;
  const int kvh = h >> 1;  // GQA group = 2
  const int tid = threadIdx.x, wave = tid >> 5, lane = tid & 31;

  const _Float16* Qbase = Qh + (((size_t)b * H_ + h) * S_ + (size_t)qt * 64) * D_;
  const _Float16* Kbase = Kh + ((size_t)b * KV_ + kvh) * S_ * D_;
  const _Float16* Vbase = Vh + ((size_t)b * KV_ + kvh) * S_ * D_;

  __shared__ _Float16 Ks[64 * D_];     // 16 KB: K tile, [k][d]
  __shared__ _Float16 Vts[D_ * 64];    // 16 KB: V tile transposed, [d][k]
  __shared__ _Float16 Ps[4][16 * 64];  //  8 KB: per-wave probs, [row][k]

  // Q fragments for this wave's 16 rows: 4 K-steps of 32 along D=128.
  half16 qf[4];
#pragma unroll
  for (int kk = 0; kk < 4; kk++)
    qf[kk] = frag16(Qbase + (size_t)(wave * 16) * D_ + kk * 32, D_, lane);

  float8 o[8];
#pragma unroll
  for (int jb = 0; jb < 8; jb++) o[jb] = zero8();
  float m[8], l[8];
#pragma unroll
  for (int e = 0; e < 8; e++) { m[e] = -1e30f; l[e] = 0.f; }

  const int lr = (lane & 16) ? 8 : 0;
  const int cb = lane & 15;
  const int r  = tid >> 1;   // staging row 0..63
  const int hh = tid & 1;    // which 64-wide d half

  for (int kt = 0; kt <= qt; kt++) {
    __syncthreads();
    {  // stage K tile async (row-major) and V tile (transposed) into LDS
      const _Float16* sk = Kbase + ((size_t)kt * 64 + r) * D_ + hh * 64;
      _Float16* dk = Ks + r * D_ + hh * 64;
#pragma unroll
      for (int i = 0; i < 8; i++) cp16_g2lds(sk + i * 8, dk + i * 8);

      _Float16 tmp[64];
      const u32x4* sv = (const u32x4*)(Vbase + ((size_t)kt * 64 + r) * D_ + hh * 64);
      u32x4* tv = (u32x4*)tmp;
#pragma unroll
      for (int i = 0; i < 8; i++) tv[i] = sv[i];
#pragma unroll
      for (int d = 0; d < 64; d++) Vts[(hh * 64 + d) * 64 + r] = tmp[d];
    }
    wait_async_lds();
    __syncthreads();

    // scores: S(16x64) = Q(16x128) * K^T; B-fragments come straight from
    // row-major Ks since lane = column-of-S == row-of-K.
    float8 sc[4];
#pragma unroll
    for (int j = 0; j < 4; j++) sc[j] = zero8();
#pragma unroll
    for (int kk = 0; kk < 4; kk++) {
#pragma unroll
      for (int j = 0; j < 4; j++) {
        half16 bf = frag16(Ks + (size_t)(j * 16) * D_ + kk * 32, D_, lane);
        sc[j] = wmma_f16(qf[kk], bf, sc[j]);
      }
    }

    // causal mask
#pragma unroll
    for (int j = 0; j < 4; j++)
#pragma unroll
      for (int e = 0; e < 8; e++) {
        int qi = qt * 64 + wave * 16 + lr + e;
        int ki = kt * 64 + j * 16 + cb;
        if (ki > qi) sc[j][e] = -1e30f;
      }

    // online softmax (per-row state replicated across each 16-lane half)
#pragma unroll
    for (int e = 0; e < 8; e++) {
      float rv = fmaxf(fmaxf(sc[0][e], sc[1][e]), fmaxf(sc[2][e], sc[3][e]));
#pragma unroll
      for (int off = 1; off < 16; off <<= 1) rv = fmaxf(rv, __shfl_xor(rv, off));
      float mn = fmaxf(m[e], rv);
      float corr = __expf(m[e] - mn);
      m[e] = mn;
      l[e] *= corr;
#pragma unroll
      for (int jb = 0; jb < 8; jb++) o[jb][e] *= corr;
      float psum = 0.f;
#pragma unroll
      for (int j = 0; j < 4; j++) {
        float p = __expf(sc[j][e] - mn);
        psum += p;
        Ps[wave][(lr + e) * 64 + j * 16 + cb] = (_Float16)p;  // C-layout -> LDS
      }
#pragma unroll
      for (int off = 1; off < 16; off <<= 1) psum += __shfl_xor(psum, off);
      l[e] += psum;
    }

    // O += P(16x64) * V(64x128); LDS ops are in-order per wave, so the
    // A-fragment reads below observe this wave's Ps stores.
#pragma unroll
    for (int kk = 0; kk < 2; kk++) {
      half16 pf = frag16(&Ps[wave][kk * 32], 64, lane);
#pragma unroll
      for (int jb = 0; jb < 8; jb++) {
        half16 vf = frag16(Vts + (size_t)(jb * 16) * 64 + kk * 32, 64, lane);
        o[jb] = wmma_f16(pf, vf, o[jb]);
      }
    }
  }

  // epilogue: divide by l, write f16 attn output in [b][s][h*D+d] for GEMM2
#pragma unroll
  for (int e = 0; e < 8; e++) {
    int qi = qt * 64 + wave * 16 + lr + e;
    float inv = 1.0f / l[e];
    size_t rowoff = ((size_t)b * S_ + qi) * (size_t)(H_ * D_) + (size_t)h * D_;
#pragma unroll
    for (int jb = 0; jb < 8; jb++)
      Oout[rowoff + jb * 16 + cb] = (_Float16)(o[jb][e] * inv);
  }
}

// ---------------- host launcher ----------------
extern "C" void kernel_launch(void* const* d_in, const int* in_sizes, int n_in,
                              void* d_out, int out_size, void* d_ws, size_t ws_size,
                              hipStream_t stream) {
  const float* hidden = (const float*)d_in[0];
  const float* Wq = (const float*)d_in[1];
  const float* Wk = (const float*)d_in[2];
  const float* Wv = (const float*)d_in[3];
  const float* Wo = (const float*)d_in[4];
  const float* qnw = (const float*)d_in[5];
  const float* knw = (const float*)d_in[6];
  const int* positions = (const int*)d_in[7];
  // d_in[8] attention_mask: causal mask computed analytically in-kernel.
  float* out = (float*)d_out;

  char* ws = (char*)d_ws;
  size_t off = 0;
  auto alloc = [&](size_t bytes) {
    void* p = ws + off;
    off += (bytes + 255) & ~(size_t)255;
    return p;
  };
  _Float16* Xh    = (_Float16*)alloc((size_t)MROWS * HID_ * 2);        // 16 MB
  _Float16* WqkvT = (_Float16*)alloc((size_t)NQKV * HID_ * 2);         // 16 MB
  _Float16* WoT   = (_Float16*)alloc((size_t)HID_ * (H_ * D_) * 2);    //  8 MB
  float*    QKV   = (float*)alloc((size_t)MROWS * NQKV * 4);           // 64 MB
  _Float16* Qh    = (_Float16*)alloc((size_t)B_ * H_ * S_ * D_ * 2);   // 16 MB
  _Float16* Kh    = (_Float16*)alloc((size_t)B_ * KV_ * S_ * D_ * 2);  //  8 MB
  _Float16* Vh    = (_Float16*)alloc((size_t)B_ * KV_ * S_ * D_ * 2);  //  8 MB
  _Float16* AttnH = (_Float16*)alloc((size_t)MROWS * H_ * D_ * 2);     // 16 MB
  (void)ws_size; (void)n_in; (void)in_sizes; (void)out_size;

  // 1) precision conversion + weight transposes
  {
    size_t n = (size_t)MROWS * HID_;
    cvt_f32_f16<<<(unsigned)(n / 4 / 256), 256, 0, stream>>>(hidden, Xh, n);
  }
  {
    size_t n = (size_t)HID_ * (H_ * D_);
    transpose_cvt<<<(unsigned)((n + 255) / 256), 256, 0, stream>>>(
        Wq, WqkvT, HID_, H_ * D_);
  }
  {
    size_t n = (size_t)HID_ * (KV_ * D_);
    transpose_cvt<<<(unsigned)((n + 255) / 256), 256, 0, stream>>>(
        Wk, WqkvT + (size_t)(H_ * D_) * HID_, HID_, KV_ * D_);
    transpose_cvt<<<(unsigned)((n + 255) / 256), 256, 0, stream>>>(
        Wv, WqkvT + (size_t)(H_ * D_ + KV_ * D_) * HID_, HID_, KV_ * D_);
  }
  {
    size_t n = (size_t)(H_ * D_) * HID_;
    transpose_cvt<<<(unsigned)((n + 255) / 256), 256, 0, stream>>>(
        Wo, WoT, H_ * D_, HID_);
  }

  // 2) QKV projection: [4096 x 2048] x [2048 x 4096] (WMMA f16)
  gemm_f16_wmma<<<dim3(NQKV / 128, MROWS / 128), 256, 0, stream>>>(
      Xh, WqkvT, QKV, MROWS, NQKV, HID_);

  // 3) RMSNorm + RoPE + repack
  qkv_post<<<dim3(S_, B_), 128, 0, stream>>>(QKV, qnw, knw, positions, Qh, Kh, Vh);

  // 4) causal flash attention (WMMA for QK^T and PV)
  flash_attn<<<dim3(S_ / 64, H_, B_), 128, 0, stream>>>(Qh, Kh, Vh, AttnH);

  // 5) output projection: [4096 x 2048] x [2048 x 2048] -> f32 d_out
  gemm_f16_wmma<<<dim3(HID_ / 128, MROWS / 128), 256, 0, stream>>>(
      AttnH, WoT, out, MROWS, HID_, H_ * D_);
}
